// BasicBiDGCNN_43911745634537
// MI455X (gfx1250) — matrix operand order repaired and checked by reading, not compiled, not run on hardware
//
#include <hip/hip_runtime.h>
#include <hip/hip_bf16.h>

typedef __attribute__((ext_vector_type(16))) _Float16 v16h;
typedef __attribute__((ext_vector_type(8)))  float    v8f;
typedef __attribute__((ext_vector_type(4)))  int      v4i;

#define BB 32
#define PP 1024
#define KK 20

#if __has_builtin(__builtin_amdgcn_global_load_async_to_lds_b128) && \
    __has_builtin(__builtin_amdgcn_s_wait_asynccnt)
#define HAVE_ASYNC_LDS 1
typedef __attribute__((address_space(1))) v4i gv4i;   // global
typedef __attribute__((address_space(3))) v4i lv4i;   // LDS
#else
#define HAVE_ASYNC_LDS 0
#endif

__device__ __forceinline__ int lane_id() { return threadIdx.x & 31; }

// Cooperative global->LDS copy; uses CDNA5 async-LDS path when available.
// nfloats must be a multiple of 4*nthreads; pointers 16B-aligned.
__device__ __forceinline__ void copy_to_lds_f32(float* lds, const float* gsrc,
                                                int nfloats, int nthreads) {
#if HAVE_ASYNC_LDS
  gv4i* g = (gv4i*)gsrc;   // C-style cast: generic->AS1 + drops const
  lv4i* s = (lv4i*)lds;    // generic->AS3
  int nvec = nfloats >> 2;
  for (int i = threadIdx.x; i < nvec; i += nthreads)
    __builtin_amdgcn_global_load_async_to_lds_b128(g + i, s + i, 0, 0);
#else
  for (int i = threadIdx.x; i < nfloats; i += nthreads) lds[i] = gsrc[i];
#endif
}

__device__ __forceinline__ void copy_to_lds_wait() {
#if HAVE_ASYNC_LDS
  __builtin_amdgcn_s_wait_asynccnt(0);
#endif
}

// ---- WMMA fragment helpers (layouts per CDNA5 ISA §7.12.2) ----

// A fragment: 16x32 f16 tile from row-major LDS buffer (leading dim ld), k offset kc.
// lane L: row = L%16; halves 0..7 -> K = kb..kb+7, halves 8..15 -> K = kb+16..kb+23,
// where kb = kc + 8*(L/16). Two contiguous 16B runs -> ds_load_b128 x2.
__device__ __forceinline__ v16h load_A(const _Float16* a, int ld, int kc) {
  int l = lane_id();
  int r = l & 15;
  int kb = kc + ((l >> 4) << 3);
  v16h o;
#pragma unroll
  for (int i = 0; i < 8; ++i) {
    o[i]     = a[r * ld + kb + i];
    o[i + 8] = a[r * ld + kb + 16 + i];
  }
  return o;
}

// Swizzled-B staging: weight element (k, n) of a Kp x N tile set goes to
//   tile = (k/32)*(N/16) + (n/16)
//   lane = 16*((k%16)/8) + n%16
//   slot = k%8 + 8*((k%32)/16)
// so each lane's 16 halves for one 32x16 B tile are contiguous (32 bytes).
// Optional biasRow is staged as K-row srcK (for folding bias into padded K).
__device__ __forceinline__ void stage_B(_Float16* dst, const float* __restrict__ src,
                                        int srcK, int srcLd, int colOff,
                                        int Kp, int nshift, int nthreads,
                                        const float* __restrict__ biasRow = nullptr) {
  int N = 1 << nshift;
  int ntiles = N >> 4;
  int total = Kp << nshift;
  for (int i = threadIdx.x; i < total; i += nthreads) {
    int k = i >> nshift;
    int n = i & (N - 1);
    float v;
    if (k < srcK)                      v = src[(size_t)k * srcLd + colOff + n];
    else if (biasRow && k == srcK)     v = biasRow[colOff + n];
    else                               v = 0.f;
    int tile = (k >> 5) * ntiles + (n >> 4);
    int lane = (((k >> 3) & 1) << 4) | (n & 15);
    int slot = (k & 7) | (((k >> 4) & 1) << 3);
    dst[(tile << 9) + (lane << 4) + slot] = (_Float16)v;
  }
}

// B fragment load from swizzled storage: one contiguous v16h per lane.
__device__ __forceinline__ v16h load_B_sw(const _Float16* w, int tile) {
  return *(const v16h*)(w + (tile << 9) + (lane_id() << 4));
}

// C initialized with per-column bias broadcast (lane L -> col nc + L%16, all 8 rows).
__device__ __forceinline__ v8f bias_C(const float* b, int nc) {
  float bv = b[nc + (lane_id() & 15)];
  v8f c;
#pragma unroll
  for (int j = 0; j < 8; ++j) c[j] = bv;
  return c;
}

__device__ __forceinline__ v8f zero_C() {
  v8f c;
#pragma unroll
  for (int j = 0; j < 8; ++j) c[j] = 0.f;
  return c;
}

__device__ __forceinline__ v8f wmma_f16(v16h a, v16h b, v8f c) {
  return __builtin_amdgcn_wmma_f32_16x16x32_f16(false, a, false, b, (short)0, c,
                                                false, false);
}

// D tile -> f16 LDS, row-major; lane L holds col L%16, rows j + 8*(L/16).
__device__ __forceinline__ void store_D(_Float16* s, int ld, int nc, v8f d, bool relu) {
  int l = lane_id();
  int n = nc + (l & 15);
  int mb = (l >> 4) << 3;
#pragma unroll
  for (int j = 0; j < 8; ++j) {
    float v = d[j];
    if (relu && v < 0.f) v = 0.f;
    s[(mb + j) * ld + n] = (_Float16)v;
  }
}

// Valid only for v >= 0 (post-ReLU): IEEE positive floats order as uints.
__device__ __forceinline__ void atomic_max_pos(float* p, float v) {
  atomicMax((unsigned int*)p, __float_as_uint(v));
}

// ---------------------------------------------------------------------------
__global__ __launch_bounds__(256) void zero_kernel(float* __restrict__ p, int n) {
  int i = blockIdx.x * 256 + threadIdx.x;
  if (i < n) p[i] = 0.f;
}

// kNN on 3-d points. One block = 256 queries of one batch; all 1024 batch points in LDS.
__global__ __launch_bounds__(256) void knn3_kernel(const float* __restrict__ pos,
                                                   int* __restrict__ idxo) {
  __shared__ alignas(16) float sx[PP * 3];
  int b = blockIdx.x >> 2;
  int q = ((blockIdx.x & 3) << 8) + threadIdx.x;
  copy_to_lds_f32(sx, pos + (size_t)b * PP * 3, PP * 3, 256);
  copy_to_lds_wait();
  __syncthreads();
  float qx = sx[q * 3 + 0], qy = sx[q * 3 + 1], qz = sx[q * 3 + 2];
  float dist[KK];
  int nid[KK];
#pragma unroll
  for (int t = 0; t < KK; ++t) { dist[t] = 3.4e38f; nid[t] = 0; }
  for (int j = 0; j < PP; ++j) {
    float dx = sx[j * 3 + 0] - qx;
    float dy = sx[j * 3 + 1] - qy;
    float dz = sx[j * 3 + 2] - qz;
    float d = dx * dx + dy * dy + dz * dz;
    if (d < dist[KK - 1]) {   // strict < keeps earliest index on ties (top_k tie rule)
      dist[KK - 1] = d; nid[KK - 1] = j;
#pragma unroll
      for (int s = KK - 1; s > 0; --s) {
        if (dist[s] < dist[s - 1]) {
          float td = dist[s]; dist[s] = dist[s - 1]; dist[s - 1] = td;
          int ti = nid[s]; nid[s] = nid[s - 1]; nid[s - 1] = ti;
        }
      }
    }
  }
  int base = (b * PP + q) * KK;
#pragma unroll
  for (int t = 0; t < KK; ++t) idxo[base + t] = nid[t];
}

// kNN on 64-d features. Candidates tiled through LDS in chunks of 128 (32KB).
__global__ __launch_bounds__(256) void knn64_kernel(const float* __restrict__ x,
                                                    int* __restrict__ idxo) {
  __shared__ alignas(16) float sf[128 * 64];
  int b = blockIdx.x >> 2;
  int q = ((blockIdx.x & 3) << 8) + threadIdx.x;
  float qf[64];
#pragma unroll
  for (int i = 0; i < 64; ++i) qf[i] = x[(size_t)(b * PP + q) * 64 + i];
  float dist[KK];
  int nid[KK];
#pragma unroll
  for (int t = 0; t < KK; ++t) { dist[t] = 3.4e38f; nid[t] = 0; }
  for (int ch = 0; ch < 8; ++ch) {
    int jb = ch * 128;
    __syncthreads();
    copy_to_lds_f32(sf, x + (size_t)(b * PP + jb) * 64, 128 * 64, 256);
    copy_to_lds_wait();
    __syncthreads();
    for (int jj = 0; jj < 128; ++jj) {
      float d = 0.f;
#pragma unroll
      for (int i = 0; i < 64; ++i) {
        float t = sf[jj * 64 + i] - qf[i];
        d += t * t;
      }
      int j = jb + jj;
      if (d < dist[KK - 1]) {
        dist[KK - 1] = d; nid[KK - 1] = j;
#pragma unroll
        for (int s = KK - 1; s > 0; --s) {
          if (dist[s] < dist[s - 1]) {
            float td = dist[s]; dist[s] = dist[s - 1]; dist[s - 1] = td;
            int ti = nid[s]; nid[s] = nid[s - 1]; nid[s - 1] = ti;
          }
        }
      }
    }
  }
  int base = (b * PP + q) * KK;
#pragma unroll
  for (int t = 0; t < KK; ++t) idxo[base + t] = nid[t];
}

// EdgeConv1: rows = (point,neighbor) edges; MLP 6(pad32)->64 ->64 ->64, then
// max over k via atomicMax into x1. 4 waves/block, 16 edge-rows per wave.
// Layer-1 A is built in registers; its bias is folded in as K-row 6 (A slot 6 = 1.0),
// so layer-1 C is the inline-constant zero.
__global__ __launch_bounds__(128) void edgeconv1_kernel(
    const float* __restrict__ pos, const int* __restrict__ idx,
    const float* __restrict__ w0, const float* __restrict__ b0,
    const float* __restrict__ w1, const float* __restrict__ b1,
    const float* __restrict__ w2, const float* __restrict__ b2,
    float* __restrict__ x1) {
  __shared__ alignas(32) _Float16 sW0[4 * 512];    // 4 B-tiles  (32x64, swizzled)
  __shared__ alignas(32) _Float16 sW1[8 * 512];    // 8 B-tiles  (64x64)
  __shared__ alignas(32) _Float16 sW2[8 * 512];
  __shared__ float sB1[64], sB2[64];
  __shared__ _Float16 sSt[4][16 * 64];             // per-wave transpose buffer
  int tid = threadIdx.x;
  __builtin_prefetch(w0, 0, 3);
  __builtin_prefetch(w1, 0, 3);
  __builtin_prefetch(w2, 0, 3);
  stage_B(sW0, w0, 6, 64, 0, 32, 6, 128, b0);      // bias -> K row 6
  stage_B(sW1, w1, 64, 64, 0, 64, 6, 128);
  stage_B(sW2, w2, 64, 64, 0, 64, 6, 128);
  if (tid < 64) { sB1[tid] = b1[tid]; sB2[tid] = b2[tid]; }
  int w = tid >> 5, l = tid & 31;
  int rowbase = blockIdx.x * 64 + w * 16;
  // build layer-1 A fragment in registers (slot 6 = 1.0 multiplies the bias row)
  v16h a0;
#pragma unroll
  for (int i = 0; i < 16; ++i) a0[i] = (_Float16)0.f;
  if (l < 16) {
    int gr = rowbase + l;
    int pt = gr / KK;
    int bbch = pt >> 10;
    int nb = idx[gr];
    const float* xi = pos + (size_t)pt * 3;
    const float* xj = pos + ((size_t)(bbch << 10) + nb) * 3;
    float x0 = xi[0], y0 = xi[1], z0 = xi[2];
    a0[0] = (_Float16)x0;
    a0[1] = (_Float16)y0;
    a0[2] = (_Float16)z0;
    a0[3] = (_Float16)(xj[0] - x0);
    a0[4] = (_Float16)(xj[1] - y0);
    a0[5] = (_Float16)(xj[2] - z0);
    a0[6] = (_Float16)1.f;
  }
  __syncthreads();   // weights staged
  _Float16* st = sSt[w];
  // layer 1 (no activation); per-wave LDS ops are in-order -> no barriers below
  v8f h[4];
#pragma unroll
  for (int nt = 0; nt < 4; ++nt)
    h[nt] = wmma_f16(a0, load_B_sw(sW0, nt), zero_C());
#pragma unroll
  for (int nt = 0; nt < 4; ++nt) store_D(st, 64, nt * 16, h[nt], false);
  // layer 2 (relu)
  v16h ax = load_A(st, 64, 0);
  v16h ay = load_A(st, 64, 32);
#pragma unroll
  for (int nt = 0; nt < 4; ++nt) {
    v8f c = bias_C(sB1, nt * 16);
    c = wmma_f16(ax, load_B_sw(sW1, nt), c);
    h[nt] = wmma_f16(ay, load_B_sw(sW1, 4 + nt), c);
  }
#pragma unroll
  for (int nt = 0; nt < 4; ++nt) store_D(st, 64, nt * 16, h[nt], true);
  // layer 3 (relu) + max over neighbors
  ax = load_A(st, 64, 0);
  ay = load_A(st, 64, 32);
  int n0 = l & 15, mb = (l >> 4) << 3;
#pragma unroll
  for (int nt = 0; nt < 4; ++nt) {
    v8f c = bias_C(sB2, nt * 16);
    c = wmma_f16(ax, load_B_sw(sW2, nt), c);
    c = wmma_f16(ay, load_B_sw(sW2, 4 + nt), c);
#pragma unroll
    for (int j = 0; j < 8; ++j) {
      float v = c[j] < 0.f ? 0.f : c[j];
      int gr = rowbase + mb + j;
      int pt = gr / KK;
      atomic_max_pos(x1 + (size_t)pt * 64 + nt * 16 + n0, v);
    }
  }
}

// EdgeConv2: 128 -> 128, relu, max over k into x2.
// A fragments gathered from global in fragment order (contiguous 8-float runs).
__global__ __launch_bounds__(128) void edgeconv2_kernel(
    const float* __restrict__ x1, const int* __restrict__ idx,
    const float* __restrict__ w0, const float* __restrict__ b0,
    float* __restrict__ x2) {
  __shared__ alignas(32) _Float16 sW[32 * 512];    // 32 B-tiles (128x128), 32KB
  __shared__ float sB[128];
  int tid = threadIdx.x;
  __builtin_prefetch(w0, 0, 3);
  stage_B(sW, w0, 128, 128, 0, 128, 7, 128);
  if (tid < 128) sB[tid] = b0[tid];
  int w = tid >> 5, l = tid & 31;
  int rowbase = blockIdx.x * 64 + w * 16;
  int r = l & 15;
  int gr0 = rowbase + r;
  int pt0 = gr0 / KK;
  int bbch = pt0 >> 10;
  int nb = idx[gr0];
  const float* xi = x1 + (size_t)pt0 * 64;
  const float* xj = x1 + ((size_t)(bbch << 10) + nb) * 64;
  v16h a[4];
#pragma unroll
  for (int kc = 0; kc < 4; ++kc) {
    int kb = kc * 32 + ((l >> 4) << 3);
#pragma unroll
    for (int i = 0; i < 8; ++i) {
      int f0 = kb + i, f1 = kb + 16 + i;
      a[kc][i]     = (_Float16)(f0 < 64 ? xi[f0] : (xj[f0 - 64] - xi[f0 - 64]));
      a[kc][i + 8] = (_Float16)(f1 < 64 ? xi[f1] : (xj[f1 - 64] - xi[f1 - 64]));
    }
  }
  __syncthreads();   // weights staged
  int n0 = l & 15, mb = (l >> 4) << 3;
#pragma unroll
  for (int nt = 0; nt < 8; ++nt) {
    v8f c = bias_C(sB, nt * 16);
#pragma unroll
    for (int kc = 0; kc < 4; ++kc)
      c = wmma_f16(a[kc], load_B_sw(sW, kc * 8 + nt), c);
#pragma unroll
    for (int j = 0; j < 8; ++j) {
      float v = c[j] < 0.f ? 0.f : c[j];
      int gr = rowbase + mb + j;
      int pt = gr / KK;
      atomic_max_pos(x2 + (size_t)pt * 128 + nt * 16 + n0, v);
    }
  }
}

// [x1|x2] (192) @ l1_w tile (192x128) + bias, relu, then max over P via atomicMax.
// grid: (row tiles 512, col tiles 8); 4 waves/block.
__global__ __launch_bounds__(128) void l1_pool_kernel(
    const float* __restrict__ x1, const float* __restrict__ x2,
    const float* __restrict__ lw, const float* __restrict__ lb,
    float* __restrict__ pooled) {
  __shared__ alignas(32) _Float16 sW[48 * 512];    // 48 B-tiles (192x128), 48KB
  __shared__ float sB[128];
  int tid = threadIdx.x;
  int colbase = blockIdx.y * 128;
  __builtin_prefetch(lw + (size_t)colbase, 0, 3);
  stage_B(sW, lw, 192, 1024, colbase, 192, 7, 128);
  if (tid < 128) sB[tid] = lb[colbase + tid];
  int w = tid >> 5, l = tid & 31;
  int rowbase = blockIdx.x * 64 + w * 16;
  int bb = rowbase >> 10;   // all 16 rows share one batch (1024 % 16 == 0)
  v16h a[6];
  {
    int r = l & 15;
    int gr = rowbase + r;
    const float* xr1 = x1 + (size_t)gr * 64;
    const float* xr2 = x2 + (size_t)gr * 128;
#pragma unroll
    for (int kc = 0; kc < 6; ++kc) {
      int kb = kc * 32 + ((l >> 4) << 3);
#pragma unroll
      for (int i = 0; i < 8; ++i) {
        int f0 = kb + i, f1 = kb + 16 + i;
        a[kc][i]     = (_Float16)(f0 < 64 ? xr1[f0] : xr2[f0 - 64]);
        a[kc][i + 8] = (_Float16)(f1 < 64 ? xr1[f1] : xr2[f1 - 64]);
      }
    }
  }
  __syncthreads();   // weights staged
  int n0 = l & 15;
#pragma unroll
  for (int nt = 0; nt < 8; ++nt) {
    v8f c = bias_C(sB, nt * 16);
#pragma unroll
    for (int kc = 0; kc < 6; ++kc)
      c = wmma_f16(a[kc], load_B_sw(sW, kc * 8 + nt), c);
    float m = 0.f;   // relu identity folded into max
#pragma unroll
    for (int j = 0; j < 8; ++j) { float v = c[j]; if (v > m) m = v; }
    atomic_max_pos(pooled + (size_t)bb * 1024 + colbase + nt * 16 + n0, m);
  }
}

// Tiny head layers: one block per row, thread-per-column striding.
__global__ __launch_bounds__(256) void dense_kernel(
    const float* __restrict__ in, const float* __restrict__ W,
    const float* __restrict__ bias, float* __restrict__ out,
    int din, int dout, int relu) {
  int row = blockIdx.x;
  const float* ir = in + (size_t)row * din;
  for (int c = threadIdx.x; c < dout; c += blockDim.x) {
    float acc = bias[c];
    for (int k = 0; k < din; ++k) acc += ir[k] * W[(size_t)k * dout + c];
    if (relu && acc < 0.f) acc = 0.f;
    out[(size_t)row * dout + c] = acc;
  }
}

__global__ __launch_bounds__(64) void logsoftmax_kernel(const float* __restrict__ in,
                                                        float* __restrict__ out) {
  int row = threadIdx.x;
  if (row >= BB) return;
  const float* r = in + row * 40;
  float mx = r[0];
  for (int i = 1; i < 40; ++i) mx = fmaxf(mx, r[i]);
  float s = 0.f;
  for (int i = 0; i < 40; ++i) s += expf(r[i] - mx);
  float ls = logf(s);
  for (int i = 0; i < 40; ++i) out[row * 40 + i] = r[i] - mx - ls;
}

extern "C" void kernel_launch(void* const* d_in, const int* in_sizes, int n_in,
                              void* d_out, int out_size, void* d_ws, size_t ws_size,
                              hipStream_t stream) {
  const float* pos   = (const float*)d_in[0];
  // d_in[1] = batch (int64) -- implicit from layout, unused
  const float* c1_w0 = (const float*)d_in[2];
  const float* c1_b0 = (const float*)d_in[3];
  const float* c1_w1 = (const float*)d_in[4];
  const float* c1_b1 = (const float*)d_in[5];
  const float* c1_w2 = (const float*)d_in[6];
  const float* c1_b2 = (const float*)d_in[7];
  const float* c2_w0 = (const float*)d_in[8];
  const float* c2_b0 = (const float*)d_in[9];
  const float* l1_w  = (const float*)d_in[10];
  const float* l1_b  = (const float*)d_in[11];
  const float* m_w0  = (const float*)d_in[12];
  const float* m_b0  = (const float*)d_in[13];
  const float* m_w1  = (const float*)d_in[14];
  const float* m_b1  = (const float*)d_in[15];
  const float* m_w2  = (const float*)d_in[16];
  const float* m_b2  = (const float*)d_in[17];

  float* ws = (float*)d_ws;
  int* idx1   = (int*)ws;                 // 655360 ints
  int* idx2   = idx1 + 655360;            // 655360 ints
  float* x1   = ws + 1310720;             // 32*1024*64
  float* x2   = x1 + 2097152;             // 32*1024*128
  float* pool = x2 + 4194304;             // 32*1024
  float* h1   = pool + 32768;             // 32*512
  float* h2   = h1 + 16384;               // 32*256
  float* lg   = h2 + 8192;                // 32*40
  float* out  = (float*)d_out;

  // zero the atomicMax accumulators (x1, x2, pool are contiguous)
  int nz = 2097152 + 4194304 + 32768;
  zero_kernel<<<(nz + 255) / 256, 256, 0, stream>>>(x1, nz);

  knn3_kernel<<<BB * 4, 256, 0, stream>>>(pos, idx1);
  edgeconv1_kernel<<<10240, 128, 0, stream>>>(pos, idx1, c1_w0, c1_b0, c1_w1, c1_b1,
                                              c1_w2, c1_b2, x1);
  knn64_kernel<<<BB * 4, 256, 0, stream>>>(x1, idx2);
  edgeconv2_kernel<<<10240, 128, 0, stream>>>(x1, idx2, c2_w0, c2_b0, x2);
  l1_pool_kernel<<<dim3(512, 8), 128, 0, stream>>>(x1, x2, l1_w, l1_b, pool);
  dense_kernel<<<BB, 256, 0, stream>>>(pool, m_w0, m_b0, h1, 1024, 512, 1);
  dense_kernel<<<BB, 256, 0, stream>>>(h1, m_w1, m_b1, h2, 512, 256, 1);
  dense_kernel<<<BB, 64, 0, stream>>>(h2, m_w2, m_b2, lg, 256, 40, 0);
  logsoftmax_kernel<<<1, 32, 0, stream>>>(lg, out);
}
